// Mamba4KT_26972394619511
// MI455X (gfx1250) — compile-verified, matrix operand order checked by reading, not run on hardware
//
#include <hip/hip_runtime.h>
#include <math.h>

// ---------------- model dims ----------------
#define QUES    1024
#define DMODEL  256
#define DINNER  512
#define DSTATE  16
#define DTRANK  16
#define DCONV   4
#define NL      2
#define BATCH   8
#define SEQL    2048
#define MROWS   (BATCH * SEQL)   // 16384

typedef __attribute__((ext_vector_type(8)))  float    v8f;
typedef __attribute__((ext_vector_type(8)))  unsigned u8v;
typedef __attribute__((ext_vector_type(16))) __bf16   bf16x16;

// ---- fp32 -> packed bf16x2 (single instruction where available) ----
__device__ __forceinline__ unsigned pack_bf16(float lo, float hi) {
#if __has_builtin(__builtin_amdgcn_cvt_pk_bf16_f32)
    typedef __attribute__((ext_vector_type(2))) __bf16 bf16x2;
    bf16x2 p = __builtin_amdgcn_cvt_pk_bf16_f32(lo, hi);
    return __builtin_bit_cast(unsigned, p);
#else
    unsigned a = __builtin_bit_cast(unsigned, lo) + 0x8000u;
    unsigned b = __builtin_bit_cast(unsigned, hi) + 0x8000u;
    return __builtin_amdgcn_perm(b, a, 0x07060302u);   // {hi[3],hi[2],lo[3],lo[2]}
#endif
}
__device__ __forceinline__ unsigned short bf16_1(float v) {
    return (unsigned short)((__builtin_bit_cast(unsigned, v) + 0x8000u) >> 16);
}

__device__ __forceinline__ float silu_f(float x) {
    return x / (1.0f + __expf(-x));
}

// ---- CDNA5 async global->LDS copy (ASYNCcnt-tracked), GV addressing ----
__device__ __forceinline__ void async_g2l_b128(const void* gsrc, void* lds_dst) {
    asm volatile("global_load_async_to_lds_b128 %0, %1, off"
                 :
                 : "v"((unsigned)(unsigned long long)lds_dst), "v"(gsrc)
                 : "memory");
}
__device__ __forceinline__ void wait_asynccnt0() {
    asm volatile("s_wait_asynccnt 0" ::: "memory");
}

// ---------------------------------------------------------------------------
// fp32 -> bf16 bulk conversion (n4 = element_count/4)
// ---------------------------------------------------------------------------
__global__ void __launch_bounds__(256)
k_cvt_bf16(const float* __restrict__ src, unsigned short* __restrict__ dst, int n4) {
    int i = blockIdx.x * 256 + threadIdx.x;
    if (i >= n4) return;
    float4 f = ((const float4*)src)[i];
    ((uint2*)dst)[i] = make_uint2(pack_bf16(f.x, f.y), pack_bf16(f.z, f.w));
}

// ---------------------------------------------------------------------------
// Embedding gather: h[m,:] = embed[x[m],:]  (fp32 residual stream + bf16 copy)
// ---------------------------------------------------------------------------
__global__ void __launch_bounds__(DMODEL)
k_embed(const int* __restrict__ x, const float* __restrict__ embed,
        float* __restrict__ h, unsigned short* __restrict__ hb) {
    int m = blockIdx.x;
    int t = threadIdx.x;
    float v = embed[(size_t)x[m] * DMODEL + t];
    h [(size_t)m * DMODEL + t] = v;
    hb[(size_t)m * DMODEL + t] = bf16_1(v);
}

// ---------------------------------------------------------------------------
// FULL bf16 WMMA GEMM: C = act(A[M,K](bf16) * W[N,K](bf16)^T + bias)
// Requires M%128==0, N%64==0, K%32==0.
// 256 threads = 8 wave32s; tile 128x64; K-step 32; wave = 32x32 (4 WMMAs).
// Tiles stream global->LDS via global_load_async_to_lds_b128, double-buffered;
// s_wait_asynccnt 0 + workgroup barrier hand buffers over.
// ACT: 0 none, 1 exact GELU.  OMODE: 0 f32 C; 1 f32 C + bf16 Cb; 2 bf16 Cb only.
// ---------------------------------------------------------------------------
template <int ACT, int OMODE>
__global__ void __launch_bounds__(256)
k_gemm_bf16(const unsigned short* __restrict__ A, int lda,   // bf16 [M,K]
            const unsigned short* __restrict__ W,            // bf16 [N,K]
            const float* __restrict__ bias,                  // [N] or nullptr
            float* __restrict__ C, unsigned short* __restrict__ Cb, int ldc,
            int M, int N, int K) {
    __shared__ __align__(16) unsigned sA[2][128][20];  // 16 pairs used; 80B rows (16B-aligned)
    __shared__ __align__(16) unsigned sB[2][64][20];

    const int tid  = threadIdx.x;
    const int lane = tid & 31;
    const int wid  = tid >> 5;
    const int wm   = wid & 3;
    const int wn   = wid >> 2;
    const int block_m = blockIdx.y * 128;
    const int block_n = blockIdx.x * 64;
    const int mrow = lane & 15;
    const int kb4  = (lane >> 4) * 4;

    v8f acc[2][2] = {};

    auto stage = [&](int bi, int kt) {
        // A tile: 128 rows x 32 bf16 = 512 x 16B chunks -> 2 per thread
#pragma unroll
        for (int i = 0; i < 2; ++i) {
            int idx = i * 256 + tid;
            int r = idx >> 2;                 // 0..127
            int c = idx & 3;                  // 16B chunk (8 bf16)
            async_g2l_b128(A + (size_t)(block_m + r) * lda + kt + c * 8,
                           &sA[bi][r][c * 4]);
        }
        // W tile: 64 rows x 32 bf16 = 256 chunks -> 1 per thread
        {
            int r = tid >> 2;                 // 0..63
            int c = tid & 3;
            async_g2l_b128(W + (size_t)(block_n + r) * K + kt + c * 8,
                           &sB[bi][r][c * 4]);
        }
    };

    stage(0, 0);
    const int nk = K >> 5;
    for (int t = 0; t < nk; ++t) {
        wait_asynccnt0();        // this wave's copies for buffer t are complete
        __syncthreads();         // publish to all waves; prior readers of buf t+1 retired
        if (t + 1 < nk) stage((t + 1) & 1, (t + 1) * 32);

        const int bi = t & 1;
        u8v au0, au1, bu0, bu1;
#pragma unroll
        for (int v = 0; v < 8; ++v) {
            int p = ((v < 4) ? v : v + 4) + kb4;
            au0[v] = sA[bi][wm * 32 +      mrow][p];
            au1[v] = sA[bi][wm * 32 + 16 + mrow][p];
            bu0[v] = sB[bi][wn * 32 +      mrow][p];
            bu1[v] = sB[bi][wn * 32 + 16 + mrow][p];
        }
        bf16x16 a0 = __builtin_bit_cast(bf16x16, au0);
        bf16x16 a1 = __builtin_bit_cast(bf16x16, au1);
        bf16x16 b0 = __builtin_bit_cast(bf16x16, bu0);
        bf16x16 b1 = __builtin_bit_cast(bf16x16, bu1);

        acc[0][0] = __builtin_amdgcn_wmma_f32_16x16x32_bf16(
            false, a0, false, b0, (short)0, acc[0][0], false, false);
        acc[0][1] = __builtin_amdgcn_wmma_f32_16x16x32_bf16(
            false, a0, false, b1, (short)0, acc[0][1], false, false);
        acc[1][0] = __builtin_amdgcn_wmma_f32_16x16x32_bf16(
            false, a1, false, b0, (short)0, acc[1][0], false, false);
        acc[1][1] = __builtin_amdgcn_wmma_f32_16x16x32_bf16(
            false, a1, false, b1, (short)0, acc[1][1], false, false);
    }

    // epilogue (C/D layout: VGPR r -> M=r, +8 for upper half-wave)
#pragma unroll
    for (int i = 0; i < 2; ++i) {
#pragma unroll
        for (int j = 0; j < 2; ++j) {
            int colg = block_n + wn * 32 + j * 16 + (lane & 15);
            float bval = bias ? bias[colg] : 0.0f;
#pragma unroll
            for (int r = 0; r < 8; ++r) {
                int rowg = block_m + wm * 32 + i * 16 + r + (lane >> 4) * 8;
                float v = acc[i][j][r] + bval;
                if (ACT == 1) v = 0.5f * v * (1.0f + erff(v * 0.70710678118654752f));
                if (OMODE != 2) C [(size_t)rowg * ldc + colg] = v;
                if (OMODE >= 1) Cb[(size_t)rowg * ldc + colg] = bf16_1(v);
            }
        }
    }
}

// ---------------------------------------------------------------------------
// RAGGED f32-input WMMA GEMM (x_proj N=48, dt_proj K=16): clamped addresses,
// cndmask masking, convert-in-staging.  ACT: 0 none, 2 softplus.
// ---------------------------------------------------------------------------
template <int ACT>
__global__ void __launch_bounds__(256)
k_gemm_rag(const float* __restrict__ A, int lda,
           const float* __restrict__ W, const float* __restrict__ bias,
           float* __restrict__ C, int ldc, int M, int N, int K) {
    __shared__ unsigned sA[128][18];
    __shared__ unsigned sB[64][18];
    const int tid  = threadIdx.x;
    const int lane = tid & 31;
    const int wid  = tid >> 5;
    const int wm   = wid & 3;
    const int wn   = wid >> 2;
    const int block_m = blockIdx.y * 128;
    const int block_n = blockIdx.x * 64;
    const int mrow = lane & 15;
    const int kb4  = (lane >> 4) * 4;
    v8f acc[2][2] = {};

    for (int kt = 0; kt < K; kt += 32) {
#pragma unroll
        for (int i = 0; i < 8; ++i) {
            int idx = i * 256 + tid;
            int r  = idx >> 4;
            int pc = idx & 15;
            int k  = kt + pc * 2;
            int k0 = (k     < K) ? k     : (K - 1);
            int k1 = (k + 1 < K) ? k + 1 : (K - 1);
            float f0 = A[(size_t)(block_m + r) * lda + k0];
            float f1 = A[(size_t)(block_m + r) * lda + k1];
            f0 = (k     < K) ? f0 : 0.0f;
            f1 = (k + 1 < K) ? f1 : 0.0f;
            sA[r][pc] = pack_bf16(f0, f1);
        }
#pragma unroll
        for (int i = 0; i < 4; ++i) {
            int idx = i * 256 + tid;
            int r  = idx >> 4;
            int pc = idx & 15;
            int k  = kt + pc * 2;
            int gn = block_n + r;
            int cn = (gn < N) ? gn : (N - 1);
            int k0 = (k     < K) ? k     : (K - 1);
            int k1 = (k + 1 < K) ? k + 1 : (K - 1);
            float f0 = W[(size_t)cn * K + k0];
            float f1 = W[(size_t)cn * K + k1];
            f0 = (gn < N && k     < K) ? f0 : 0.0f;
            f1 = (gn < N && k + 1 < K) ? f1 : 0.0f;
            sB[r][pc] = pack_bf16(f0, f1);
        }
        __syncthreads();

        u8v au0, au1, bu0, bu1;
#pragma unroll
        for (int v = 0; v < 8; ++v) {
            int p = ((v < 4) ? v : v + 4) + kb4;
            au0[v] = sA[wm * 32 +      mrow][p];
            au1[v] = sA[wm * 32 + 16 + mrow][p];
            bu0[v] = sB[wn * 32 +      mrow][p];
            bu1[v] = sB[wn * 32 + 16 + mrow][p];
        }
        bf16x16 a0 = __builtin_bit_cast(bf16x16, au0);
        bf16x16 a1 = __builtin_bit_cast(bf16x16, au1);
        bf16x16 b0 = __builtin_bit_cast(bf16x16, bu0);
        bf16x16 b1 = __builtin_bit_cast(bf16x16, bu1);

        acc[0][0] = __builtin_amdgcn_wmma_f32_16x16x32_bf16(
            false, a0, false, b0, (short)0, acc[0][0], false, false);
        acc[0][1] = __builtin_amdgcn_wmma_f32_16x16x32_bf16(
            false, a0, false, b1, (short)0, acc[0][1], false, false);
        acc[1][0] = __builtin_amdgcn_wmma_f32_16x16x32_bf16(
            false, a1, false, b0, (short)0, acc[1][0], false, false);
        acc[1][1] = __builtin_amdgcn_wmma_f32_16x16x32_bf16(
            false, a1, false, b1, (short)0, acc[1][1], false, false);
        __syncthreads();
    }

#pragma unroll
    for (int i = 0; i < 2; ++i) {
#pragma unroll
        for (int j = 0; j < 2; ++j) {
            int colg = block_n + wn * 32 + j * 16 + (lane & 15);
            if (colg >= N) continue;
            float bval = bias ? bias[colg] : 0.0f;
#pragma unroll
            for (int r = 0; r < 8; ++r) {
                int rowg = block_m + wm * 32 + i * 16 + r + (lane >> 4) * 8;
                float v = acc[i][j][r] + bval;
                if (ACT == 2) v = (v > 20.0f) ? v : log1pf(__expf(v));
                C[(size_t)rowg * ldc + colg] = v;
            }
        }
    }
}

// ---------------------------------------------------------------------------
// Depthwise causal conv1d (kernel 4) + bias + SiLU, branchless boundary.
// ---------------------------------------------------------------------------
__global__ void __launch_bounds__(256)
k_conv_silu(const float* __restrict__ xz, const float* __restrict__ cw,
            const float* __restrict__ cb, float* __restrict__ xc) {
    int idx = blockIdx.x * 256 + threadIdx.x;
    if (idx >= MROWS * DINNER) return;
    int d = idx & (DINNER - 1);
    int m = idx >> 9;
    int l = m & (SEQL - 1);
    float acc = cb[d];
#pragma unroll
    for (int k = 0; k < DCONV; ++k) {
        int ls = l - (DCONV - 1) + k;
        int ms = (ls >= 0) ? (m - (DCONV - 1) + k) : m;
        float v = xz[(size_t)ms * (2 * DINNER) + d];
        acc += ((ls >= 0) ? v : 0.0f) * cw[d * DCONV + k];
    }
    xc[(size_t)m * DINNER + d] = silu_f(acc);
}

// ---------------------------------------------------------------------------
// Selective scan + D skip + SiLU(z) gating -> bf16 output ycb (out_proj A).
// Shared 48-float xdbl row for step l+1 async-prefetched into double-buffered
// LDS (global_load_async_to_lds_b128) while step l computes.
// ---------------------------------------------------------------------------
__global__ void __launch_bounds__(DINNER)
k_scan(const float* __restrict__ dtb, const float* __restrict__ xdbl,
       const float* __restrict__ xz, const float* __restrict__ Alog,
       const float* __restrict__ Dp, const float* __restrict__ xc,
       unsigned short* __restrict__ ycb) {
    __shared__ __align__(16) float srow[2][64];
    int b = blockIdx.x;
    int d = threadIdx.x;
    float a[DSTATE], hs[DSTATE];
#pragma unroll
    for (int s = 0; s < DSTATE; ++s) {
        a[s]  = -__expf(Alog[d * DSTATE + s]);
        hs[s] = 0.0f;
    }
    float dpar = Dp[d];
    const size_t mbase = (size_t)b * SEQL;

    if (d < 12)
        async_g2l_b128(xdbl + mbase * (DTRANK + 2 * DSTATE) + d * 4, &srow[0][d * 4]);

    for (int l = 0; l < SEQL; ++l) {
        wait_asynccnt0();
        __syncthreads();
        if (d < 12 && l + 1 < SEQL)
            async_g2l_b128(xdbl + (mbase + l + 1) * (DTRANK + 2 * DSTATE) + d * 4,
                           &srow[(l + 1) & 1][d * 4]);

        const float* row = srow[l & 1];
        size_t m = mbase + l;
        float dt = dtb[m * DINNER + d];
        float xv = xc[m * DINNER + d];
        float dtx = dt * xv;
        float y = 0.0f;
#pragma unroll
        for (int s = 0; s < DSTATE; ++s) {
            float dA = __expf(dt * a[s]);
            hs[s] = dA * hs[s] + dtx * row[DTRANK + s];
            y += hs[s] * row[DTRANK + DSTATE + s];
        }
        float z = xz[m * (2 * DINNER) + DINNER + d];
        ycb[m * DINNER + d] = bf16_1((y + dpar * xv) * silu_f(z));
    }
}

// ---------------------------------------------------------------------------
// h = LayerNorm(res + h) * g + b; also emits bf16 copy for downstream GEMM A.
// ---------------------------------------------------------------------------
__global__ void __launch_bounds__(DMODEL)
k_add_ln(const float* __restrict__ res, float* __restrict__ h,
         unsigned short* __restrict__ hb,
         const float* __restrict__ g, const float* __restrict__ b) {
    __shared__ float s1[DMODEL];
    __shared__ float s2[DMODEL];
    int row = blockIdx.x;
    int t   = threadIdx.x;
    float v = res[(size_t)row * DMODEL + t] + h[(size_t)row * DMODEL + t];
    s1[t] = v;
    s2[t] = v * v;
    __syncthreads();
    for (int off = DMODEL / 2; off > 0; off >>= 1) {
        if (t < off) { s1[t] += s1[t + off]; s2[t] += s2[t + off]; }
        __syncthreads();
    }
    float mean = s1[0] * (1.0f / DMODEL);
    float var  = s2[0] * (1.0f / DMODEL) - mean * mean;
    float inv  = rsqrtf(var + 1e-12f);
    float o = (v - mean) * inv * g[t] + b[t];
    h [(size_t)row * DMODEL + t] = o;
    hb[(size_t)row * DMODEL + t] = bf16_1(o);
}

// ---------------------------------------------------------------------------
// Host orchestration
// ---------------------------------------------------------------------------
extern "C" void kernel_launch(void* const* d_in, const int* in_sizes, int n_in,
                              void* d_out, int out_size, void* d_ws, size_t ws_size,
                              hipStream_t stream) {
    const int*   x     = (const int*)  d_in[0];
    const float* embed = (const float*)d_in[2];
    const float* ipw   = (const float*)d_in[3];
    const float* cw    = (const float*)d_in[4];
    const float* cb    = (const float*)d_in[5];
    const float* xpw   = (const float*)d_in[6];
    const float* dpw   = (const float*)d_in[7];
    const float* dpb   = (const float*)d_in[8];
    const float* alog  = (const float*)d_in[9];
    const float* dpar  = (const float*)d_in[10];
    const float* opw   = (const float*)d_in[11];
    const float* ln1g  = (const float*)d_in[12];
    const float* ln1b  = (const float*)d_in[13];
    const float* fw1   = (const float*)d_in[14];
    const float* fb1   = (const float*)d_in[15];
    const float* fw2   = (const float*)d_in[16];
    const float* fb2   = (const float*)d_in[17];
    const float* ln2g  = (const float*)d_in[18];
    const float* ln2b  = (const float*)d_in[19];
    const float* fcw   = (const float*)d_in[20];
    const float* fcb   = (const float*)d_in[21];

    // -------- workspace layout --------
    char* ws = (char*)d_ws;
    size_t off = 0;
    auto carve = [&](size_t bytes) { char* p = ws + off; off += bytes; return p; };
    float*          h    = (float*)         carve((size_t)MROWS * DMODEL * 4);       // 16 MB
    float*          xz   = (float*)         carve((size_t)MROWS * 2 * DINNER * 4);   // 64 MB
    float*          xc   = (float*)         carve((size_t)MROWS * DINNER * 4);       // 32 MB
    float*          dt   = (float*)         carve((size_t)MROWS * DINNER * 4);       // 32 MB
    float*          xdbl = (float*)         carve((size_t)MROWS * 48 * 4);           //  3 MB
    float*          mbuf = (float*)         carve((size_t)MROWS * DMODEL * 4);       // 16 MB
    unsigned short* hb   = (unsigned short*)carve((size_t)MROWS * DMODEL * 2);       //  8 MB
    unsigned short* ycb  = (unsigned short*)carve((size_t)MROWS * DINNER * 2);       // 16 MB
    unsigned short* xzb  = (unsigned short*)carve((size_t)MROWS * 4 * DMODEL * 2);   // 32 MB
    unsigned short* ipwb = (unsigned short*)carve((size_t)NL * 2 * DINNER * DMODEL * 2);
    unsigned short* opwb = (unsigned short*)carve((size_t)NL * DMODEL * DINNER * 2);
    unsigned short* fw1b = (unsigned short*)carve((size_t)NL * 4 * DMODEL * DMODEL * 2);
    unsigned short* fw2b = (unsigned short*)carve((size_t)NL * DMODEL * 4 * DMODEL * 2);
    unsigned short* fcwb = (unsigned short*)carve((size_t)QUES * DMODEL * 2);

    const int M = MROWS;
    dim3 blk(256);
    auto gemm_grid = [](int N) { return dim3((unsigned)(N / 64), (unsigned)(MROWS / 128)); };
    auto cvt = [&](const float* s, unsigned short* d, size_t n) {
        int n4 = (int)(n / 4);
        k_cvt_bf16<<<dim3((n4 + 255) / 256), blk, 0, stream>>>(s, d, n4);
    };

    // -------- one-time weight precision lowering (fp32 -> bf16) --------
    cvt(ipw, ipwb, (size_t)NL * 2 * DINNER * DMODEL);
    cvt(opw, opwb, (size_t)NL * DMODEL * DINNER);
    cvt(fw1, fw1b, (size_t)NL * 4 * DMODEL * DMODEL);
    cvt(fw2, fw2b, (size_t)NL * DMODEL * 4 * DMODEL);
    cvt(fcw, fcwb, (size_t)QUES * DMODEL);

    // h = embed[x]  (+ bf16 copy)
    k_embed<<<dim3(MROWS), dim3(DMODEL), 0, stream>>>(x, embed, h, hb);

    for (int li = 0; li < NL; ++li) {
        const unsigned short* ipwb_l = ipwb + (size_t)li * 2 * DINNER * DMODEL;
        const unsigned short* opwb_l = opwb + (size_t)li * DMODEL * DINNER;
        const unsigned short* fw1b_l = fw1b + (size_t)li * 4 * DMODEL * DMODEL;
        const unsigned short* fw2b_l = fw2b + (size_t)li * DMODEL * 4 * DMODEL;
        const float* cw_l   = cw   + (size_t)li * DINNER * DCONV;
        const float* cb_l   = cb   + (size_t)li * DINNER;
        const float* xpw_l  = xpw  + (size_t)li * 48 * DINNER;
        const float* dpw_l  = dpw  + (size_t)li * DINNER * DTRANK;
        const float* dpb_l  = dpb  + (size_t)li * DINNER;
        const float* alog_l = alog + (size_t)li * DINNER * DSTATE;
        const float* dpar_l = dpar + (size_t)li * DINNER;

        // xz = h @ in_proj^T  [M,1024]
        k_gemm_bf16<0, 0><<<gemm_grid(2 * DINNER), blk, 0, stream>>>(
            hb, DMODEL, ipwb_l, nullptr, xz, nullptr, 2 * DINNER, M, 2 * DINNER, DMODEL);

        // xc = silu(causal depthwise conv(xs) + cb)
        k_conv_silu<<<dim3((MROWS * DINNER + 255) / 256), blk, 0, stream>>>(xz, cw_l, cb_l, xc);

        // xdbl = xc @ x_proj^T  [M,48]  (ragged N)
        k_gemm_rag<0><<<dim3(1, MROWS / 128), blk, 0, stream>>>(
            xc, DINNER, xpw_l, nullptr, xdbl, 48, M, 48, DINNER);

        // dt = softplus(xdbl[:, :16] @ dt_proj^T + dt_b)  [M,512]  (ragged K=16)
        k_gemm_rag<2><<<gemm_grid(DINNER), blk, 0, stream>>>(
            xdbl, 48, dpw_l, dpb_l, dt, DINNER, M, DINNER, DTRANK);

        // selective scan + D skip + silu(z) gating -> bf16 ycb
        k_scan<<<dim3(BATCH), dim3(DINNER), 0, stream>>>(dt, xdbl, xz, alog_l, dpar_l, xc, ycb);

        // mbuf = y @ out_proj^T  [M,256]
        k_gemm_bf16<0, 0><<<gemm_grid(DMODEL), blk, 0, stream>>>(
            ycb, DINNER, opwb_l, nullptr, mbuf, nullptr, DMODEL, M, DMODEL, DINNER);

        // h = LN(mbuf + h)  (+ bf16 copy)
        k_add_ln<<<dim3(MROWS), dim3(DMODEL), 0, stream>>>(
            mbuf, h, hb, ln1g + li * DMODEL, ln1b + li * DMODEL);

        // ffn_mid(bf16 only) = gelu(h @ w1^T + b1)  [M,1024]
        k_gemm_bf16<1, 2><<<gemm_grid(4 * DMODEL), blk, 0, stream>>>(
            hb, DMODEL, fw1b_l, fb1 + (size_t)li * 4 * DMODEL,
            nullptr, xzb, 4 * DMODEL, M, 4 * DMODEL, DMODEL);

        // mbuf = ffn_mid @ w2^T + b2  [M,256]
        k_gemm_bf16<0, 0><<<gemm_grid(DMODEL), blk, 0, stream>>>(
            xzb, 4 * DMODEL, fw2b_l, fb2 + (size_t)li * DMODEL,
            mbuf, nullptr, DMODEL, M, DMODEL, 4 * DMODEL);

        // h = LN(mbuf + h)  (+ bf16 copy)
        k_add_ln<<<dim3(MROWS), dim3(DMODEL), 0, stream>>>(
            mbuf, h, hb, ln2g + li * DMODEL, ln2b + li * DMODEL);
    }

    // logits = h @ fc^T + fc_b  [M,1024] -> d_out (fp32)
    k_gemm_bf16<0, 0><<<gemm_grid(QUES), blk, 0, stream>>>(
        hb, DMODEL, fcwb, fcb, (float*)d_out, nullptr, QUES, M, QUES, DMODEL);
}
